// Model_39676907881027
// MI455X (gfx1250) — matrix operand order repaired and checked by reading, not compiled, not run on hardware
//
#include <hip/hip_runtime.h>
#include <hip/hip_bf16.h>
#include <math.h>

typedef __attribute__((ext_vector_type(16))) _Float16 v16h;
typedef __attribute__((ext_vector_type(8)))  _Float16 v8h;
typedef __attribute__((ext_vector_type(4)))  _Float16 h4;
typedef __attribute__((ext_vector_type(2)))  __fp16   fp16x2;
typedef __attribute__((ext_vector_type(8)))  float    v8f;

#define DIMM   1024
#define HEADS  16
#define HD     64
#define SEQ    2048
#define BATCH  2
#define LOG2E  1.4426950408889634f
#define CHUNK  64      // keys per attention pipeline stage
#define KSTR   72      // LDS row stride in halves (bank-conflict-free)

// Build a 16-half A/B fragment from two 16-byte LDS/global chunks.
static __device__ __forceinline__ v16h ldpair(const _Float16* p0, const _Float16* p1) {
    v8h lo = *(const v8h*)p0;
    v8h hi = *(const v8h*)p1;
    return __builtin_shufflevector(lo, hi, 0,1,2,3,4,5,6,7,8,9,10,11,12,13,14,15);
}

static __device__ __forceinline__ v8f wmma_f16(v16h a, v16h b, v8f c) {
    return __builtin_amdgcn_wmma_f32_16x16x32_f16(false, a, false, b, (short)0, c, false, false);
}

static __device__ __forceinline__ int pack2(float a, float b) {
    union { fp16x2 h; int i; } u;
    u.h = __builtin_amdgcn_cvt_pkrtz(a, b);   // v_cvt_pk_rtz_f16_f32
    return u.i;
}

// Async DMA: 16 bytes global -> LDS per lane, tracked by ASYNCcnt.
static __device__ __forceinline__ void async_b128(unsigned lds_addr,
                                                  unsigned long long gbase,
                                                  unsigned byte_off) {
    asm volatile("global_load_async_to_lds_b128 %0, %1, %2"
                 :: "v"(lds_addr), "v"(byte_off), "s"(gbase) : "memory");
}
static __device__ __forceinline__ void wait_async0() {
    asm volatile("s_wait_asynccnt 0x0" ::: "memory");
}

// P^T B-fragment build: pa/pb hold exp() of (key = base+r+8g) / (base+16+r+8g).
static __device__ __forceinline__ v16h build_pfrag(const float* pa, const float* pb, int g) {
    int pd0[4], pd1[4], px0[4], px1[4];
    #pragma unroll
    for (int j = 0; j < 4; ++j) {
        pd0[j] = pack2(pa[2 * j], pa[2 * j + 1]);
        pd1[j] = pack2(pb[2 * j], pb[2 * j + 1]);
    }
    #pragma unroll
    for (int j = 0; j < 4; ++j) {
        px0[j] = __shfl_xor(pd0[j], 16, 32);
        px1[j] = __shfl_xor(pd1[j], 16, 32);
    }
    union { int i[8]; v16h v; } bu;
    #pragma unroll
    for (int j = 0; j < 4; ++j) {
        bu.i[j]     = g ? px1[j] : pd0[j];   // keys base+16g .. +7
        bu.i[4 + j] = g ? pd1[j] : px0[j];   // keys base+16g+8 .. +15
    }
    return bu.v;
}

// ---------------------------------------------------------------------------
// Projection GEMM: Y = X * W^T + bias  ->  f16 head-major [b, h, t, d].
// ---------------------------------------------------------------------------
__global__ __launch_bounds__(256) void proj_kernel(
    const float* __restrict__ X, const float* __restrict__ W,
    const float* __restrict__ bias, _Float16* __restrict__ dst,
    const float* __restrict__ inv_scale, int is_q)
{
    __shared__ __align__(16) _Float16 As[128 * 40];
    __shared__ __align__(16) _Float16 Bs[128 * 40];

    const int tid  = threadIdx.x;
    const int lane = tid & 31;
    const int wid  = tid >> 5;
    const int wm   = wid & 1;
    const int wn   = wid >> 1;
    const int g    = lane >> 4;
    const int l16  = lane & 15;
    const int blockM = blockIdx.x * 128;
    const int blockN = blockIdx.y * 128;

    float mul = 1.0f;
    if (is_q) mul = 0.125f * LOG2E / inv_scale[0];

    v8f acc[4][2];
    #pragma unroll
    for (int i = 0; i < 4; ++i)
        #pragma unroll
        for (int j = 0; j < 2; ++j)
            acc[i][j] = (v8f){0.f,0.f,0.f,0.f,0.f,0.f,0.f,0.f};

    for (int k0 = 0; k0 < DIMM; k0 += 32) {
        {
            const int r  = tid >> 3;
            const int c4 = tid & 7;
            #pragma unroll
            for (int p = 0; p < 4; ++p) {
                const int row = p * 32 + r;
                const float* xp = X + (size_t)(blockM + row) * DIMM + k0 + c4 * 4;
                const float* wp = W + (size_t)(blockN + row) * DIMM + k0 + c4 * 4;
                float4 a = *(const float4*)xp;
                float4 b = *(const float4*)wp;
                if (k0 + 32 < DIMM) {               // prefetch next K-tile line
                    __builtin_prefetch(xp + 32, 0, 0);
                    __builtin_prefetch(wp + 32, 0, 0);
                }
                h4 ah = { (_Float16)a.x, (_Float16)a.y, (_Float16)a.z, (_Float16)a.w };
                h4 bh = { (_Float16)b.x, (_Float16)b.y, (_Float16)b.z, (_Float16)b.w };
                *(h4*)&As[row * 40 + c4 * 4] = ah;
                *(h4*)&Bs[row * 40 + c4 * 4] = bh;
            }
        }
        __syncthreads();

        v16h af[4], bf[2];
        #pragma unroll
        for (int mt = 0; mt < 4; ++mt) {
            const _Float16* p = &As[(wm * 64 + mt * 16 + l16) * 40 + g * 8];
            af[mt] = ldpair(p, p + 16);
        }
        #pragma unroll
        for (int nt = 0; nt < 2; ++nt) {
            const _Float16* p = &Bs[(wn * 32 + nt * 16 + l16) * 40 + g * 8];
            bf[nt] = ldpair(p, p + 16);
        }
        #pragma unroll
        for (int mt = 0; mt < 4; ++mt)
            #pragma unroll
            for (int nt = 0; nt < 2; ++nt)
                acc[mt][nt] = wmma_f16(af[mt], bf[nt], acc[mt][nt]);
        __syncthreads();
    }

    #pragma unroll
    for (int nt = 0; nt < 2; ++nt) {
        const int o  = blockN + wn * 32 + nt * 16 + l16;
        const float bo = bias[o];
        const int hh = o >> 6, d = o & 63;
        #pragma unroll
        for (int mt = 0; mt < 4; ++mt) {
            #pragma unroll
            for (int r = 0; r < 8; ++r) {
                const int m = blockM + wm * 64 + mt * 16 + r + 8 * g;
                const int b = m >> 11, t = m & 2047;
                const float v = (acc[mt][nt][r] + bo) * mul;
                dst[(((size_t)(b * HEADS + hh) * SEQ + t) * HD + d)] = (_Float16)v;
            }
        }
    }
}

// ---------------------------------------------------------------------------
// Flash attention, 64-key software-pipelined chunks:
//   K staged via global_load_async_to_lds_b128 (double-buffered LDS),
//   V loaded early to VGPRs and stored transposed, one barrier per chunk.
// ---------------------------------------------------------------------------
__global__ __launch_bounds__(256) void attn_kernel(
    const _Float16* __restrict__ q16, const _Float16* __restrict__ k16,
    const _Float16* __restrict__ v16, float* __restrict__ out)
{
    __shared__ __align__(16) _Float16 Ks[2][CHUNK * KSTR];  // [key][d]
    __shared__ __align__(16) _Float16 Vt[2][HD * KSTR];     // [d][key]

    const int tid  = threadIdx.x;
    const int lane = tid & 31;
    const int wid  = tid >> 5;
    const int g    = lane >> 4;
    const int l16  = lane & 15;
    const int bh   = blockIdx.y;
    const int b    = bh >> 4, h = bh & 15;
    const int qrow = blockIdx.x * 128 + wid * 16 + l16;
    const unsigned head_elem = (unsigned)bh * SEQ * HD;

    // Cooperative-load mapping: 64 rows x 64 d, 16 halves (2 x b128) per thread.
    const int row = tid >> 2;          // key row 0..63
    const int dg  = (tid & 3) * 16;    // d start
    const unsigned long long kbase = (unsigned long long)(uintptr_t)k16;
    unsigned ldsK[2];
    ldsK[0] = (unsigned)(uintptr_t)&Ks[0][row * KSTR + dg];
    ldsK[1] = (unsigned)(uintptr_t)&Ks[1][row * KSTR + dg];

    // Q^T B-fragments (N = query = l16, K = d), loaded once.
    v16h qf[2];
    #pragma unroll
    for (int dh = 0; dh < 2; ++dh) {
        const _Float16* p = q16 + (size_t)head_elem + (size_t)qrow * HD + dh * 32 + g * 8;
        qf[dh] = ldpair(p, p + 16);
    }

    v8f Ot[4];
    #pragma unroll
    for (int i = 0; i < 4; ++i) Ot[i] = (v8f){0.f,0.f,0.f,0.f,0.f,0.f,0.f,0.f};
    float mrun = -INFINITY, lrun = 0.f;

    // Prologue: stage chunk 0 into buffer 0.
    {
        const unsigned goff = 2u * (head_elem + (unsigned)row * HD + dg);
        async_b128(ldsK[0], kbase, goff);
        async_b128(ldsK[0] + 16, kbase, goff + 16);
        const _Float16* vp = v16 + head_elem + (size_t)row * HD + dg;
        v8h va = *(const v8h*)vp;
        v8h vb = *(const v8h*)(vp + 8);
        #pragma unroll
        for (int i = 0; i < 8; ++i) {
            Vt[0][(dg + i) * KSTR + row]     = va[i];
            Vt[0][(dg + 8 + i) * KSTR + row] = vb[i];
        }
        wait_async0();
    }
    __syncthreads();

    int buf = 0;
    for (int c0 = 0; c0 < SEQ; c0 += CHUNK) {
        const int nxt = buf ^ 1;
        const bool more = (c0 + CHUNK) < SEQ;
        v8h va, vb;
        if (more) {   // issue next chunk's K async + V global loads early
            const unsigned goff = 2u * (head_elem + (unsigned)(c0 + CHUNK + row) * HD + dg);
            async_b128(ldsK[nxt], kbase, goff);
            async_b128(ldsK[nxt] + 16, kbase, goff + 16);
            const _Float16* vp = v16 + head_elem + (size_t)(c0 + CHUNK + row) * HD + dg;
            va = *(const v8h*)vp;
            vb = *(const v8h*)(vp + 8);
        }

        // ---- scores: S^T tiles, st[kt] = (key = 16kt + r + 8g, query = l16)
        v8f st[4];
        #pragma unroll
        for (int kt = 0; kt < 4; ++kt) {
            v8f s = (v8f){0.f,0.f,0.f,0.f,0.f,0.f,0.f,0.f};
            #pragma unroll
            for (int dh = 0; dh < 2; ++dh) {
                const _Float16* p = &Ks[buf][(kt * 16 + l16) * KSTR + dh * 32 + g * 8];
                s = wmma_f16(ldpair(p, p + 16), qf[dh], s);
            }
            st[kt] = s;
        }

        // ---- online softmax over 64 keys (base-2)
        float mloc = st[0][0];
        #pragma unroll
        for (int kt = 0; kt < 4; ++kt)
            #pragma unroll
            for (int r = 0; r < 8; ++r) mloc = fmaxf(mloc, st[kt][r]);
        mloc = fmaxf(mloc, __shfl_xor(mloc, 16, 32));
        const float mnew = fmaxf(mrun, mloc);
        const float corr = exp2f(mrun - mnew);
        float pv[4][8], psum = 0.f;
        #pragma unroll
        for (int kt = 0; kt < 4; ++kt)
            #pragma unroll
            for (int r = 0; r < 8; ++r) { pv[kt][r] = exp2f(st[kt][r] - mnew); psum += pv[kt][r]; }
        psum += __shfl_xor(psum, 16, 32);
        lrun = lrun * corr + psum;
        mrun = mnew;
        #pragma unroll
        for (int mt = 0; mt < 4; ++mt)
            #pragma unroll
            for (int r = 0; r < 8; ++r) Ot[mt][r] *= corr;

        // ---- P^T B-fragments for keys 0..31 and 32..63
        const v16h bp0 = build_pfrag(pv[0], pv[1], g);
        const v16h bp1 = build_pfrag(pv[2], pv[3], g);

        // ---- O^T += V^T * P^T
        #pragma unroll
        for (int mt = 0; mt < 4; ++mt) {
            const _Float16* p0 = &Vt[buf][(mt * 16 + l16) * KSTR + g * 16];
            Ot[mt] = wmma_f16(ldpair(p0, p0 + 8), bp0, Ot[mt]);
            const _Float16* p1 = &Vt[buf][(mt * 16 + l16) * KSTR + 32 + g * 16];
            Ot[mt] = wmma_f16(ldpair(p1, p1 + 8), bp1, Ot[mt]);
        }

        if (more) {   // commit next V chunk (transposed) + ensure async K landed
            #pragma unroll
            for (int i = 0; i < 8; ++i) {
                Vt[nxt][(dg + i) * KSTR + row]     = va[i];
                Vt[nxt][(dg + 8 + i) * KSTR + row] = vb[i];
            }
            wait_async0();
        }
        __syncthreads();
        buf = nxt;
    }

    // ---- normalize, write with reference permutation out[b, (t*64+d)*16 + h]
    const float inv_l = 1.0f / lrun;
    #pragma unroll
    for (int mt = 0; mt < 4; ++mt) {
        #pragma unroll
        for (int r = 0; r < 8; ++r) {
            const int d = mt * 16 + r + 8 * g;
            out[(size_t)b * (SEQ * HD * HEADS) + ((size_t)qrow * HD + d) * HEADS + h]
                = Ot[mt][r] * inv_l;
        }
    }
}

extern "C" void kernel_launch(void* const* d_in, const int* in_sizes, int n_in,
                              void* d_out, int out_size, void* d_ws, size_t ws_size,
                              hipStream_t stream) {
    const float* q   = (const float*)d_in[0];
    const float* k   = (const float*)d_in[1];
    const float* v   = (const float*)d_in[2];
    const float* inv = (const float*)d_in[4];
    const float* Wq  = (const float*)d_in[5];
    const float* bq  = (const float*)d_in[6];
    const float* Wk  = (const float*)d_in[7];
    const float* bk  = (const float*)d_in[8];
    const float* Wv  = (const float*)d_in[9];
    const float* bv  = (const float*)d_in[10];

    const size_t head_elems = (size_t)BATCH * HEADS * SEQ * HD;
    _Float16* q16 = (_Float16*)d_ws;
    _Float16* k16 = q16 + head_elems;
    _Float16* v16 = k16 + head_elems;

    dim3 pgrid((BATCH * SEQ) / 128, DIMM / 128);
    proj_kernel<<<pgrid, 256, 0, stream>>>(q, Wq, bq, q16, inv, 1);
    proj_kernel<<<pgrid, 256, 0, stream>>>(k, Wk, bk, k16, inv, 0);
    proj_kernel<<<pgrid, 256, 0, stream>>>(v, Wv, bv, v16, inv, 0);

    dim3 agrid(SEQ / 128, BATCH * HEADS);
    attn_kernel<<<agrid, 256, 0, stream>>>(q16, k16, v16, (float*)d_out);
}
